// CharRNN_57406532878348
// MI455X (gfx1250) — matrix operand order repaired and checked by reading, not compiled, not run on hardware
//
#include <hip/hip_runtime.h>
#include <hip/hip_bf16.h>

// CharRNN forward for MI455X (gfx1250): bf16 WMMA 16x16x32, f32 accum.
// Phase 1: xproj = gather(embedding, x) @ W_xh + b_h     -> ws  [S,B,H]
// Phase 2: h_t = tanh(xproj_t + h_{t-1} @ W_hh)  (batch-parallel recurrence,
//          hs overwrites xproj in place in ws; h_last -> tail of d_out)
// Phase 3: logits = hs @ W_hy + b_y                      -> d_out [B,S,V]
//
// - Weights staged in LDS pre-swizzled into per-lane WMMA B-fragment order
//   (2x ds_load_b128 per fragment).
// - Recurrence holds all 8 B fragments of W_hh in VGPRs across the entire
//   1024-step serial loop: inner loop = 16 ds_load_b128 (A) + 8 WMMA.
// - Activation staging uses float4 global loads + packed bf16x4 LDS stores (weights: scalar).

#define VOCAB  256
#define HIDDEN 256
#define EMBED  128
#define BATCH  128
#define SEQ    1024

typedef __attribute__((ext_vector_type(16))) __bf16 v16bf;
typedef __attribute__((ext_vector_type(8)))  __bf16 v8bf;
typedef __attribute__((ext_vector_type(4)))  __bf16 v4bf;
typedef __attribute__((ext_vector_type(8)))  float  v8f;

// ---- A fragment (16x32 tile, row-major source, stride ld), ISA 7.12.2 ----
// lane: M = lane&15, kbase = (lane>>4)*16 ; elems = K[kbase..kbase+15]
// -> one contiguous 32-byte run -> 2x ds_load_b128.
__device__ __forceinline__ v16bf load_a_frag(const __bf16* src, int ld, int lane) {
  const int m = lane & 15;
  const int kbase = (lane >> 4) * 16;
  const __bf16* p = src + m * ld + kbase;
  v8bf lo = *(const v8bf*)(p);
  v8bf hi = *(const v8bf*)(p + 8);
  return __builtin_shufflevector(lo, hi, 0, 1, 2, 3, 4, 5, 6, 7,
                                         8, 9, 10, 11, 12, 13, 14, 15);
}

// ---- B fragment from pre-swizzled LDS: 16 contiguous bf16 per lane ----
__device__ __forceinline__ v16bf load_b_swz(const __bf16* swz, int frag, int lane) {
  const __bf16* p = swz + ((size_t)frag * 32 + lane) * 16;
  v8bf lo = *(const v8bf*)(p);
  v8bf hi = *(const v8bf*)(p + 8);
  return __builtin_shufflevector(lo, hi, 0, 1, 2, 3, 4, 5, 6, 7,
                                         8, 9, 10, 11, 12, 13, 14, 15);
}

// float4 -> 4x bf16 packed 8-byte LDS store
__device__ __forceinline__ void cvt_store4(__bf16* dst, const float4 v) {
  v4bf o;
  o[0] = (__bf16)v.x; o[1] = (__bf16)v.y; o[2] = (__bf16)v.z; o[3] = (__bf16)v.w;
  *(v4bf*)dst = o;
}

// Cooperative fill of a swizzled B-weight buffer from a row-major f32 matrix
// W[K x N] (ld = N). frag id = kc*16 + nt. Per (frag,lane): elem e has
// K = kc*32 + (lane>>4)*16 + e, N = nt*16 + (lane&15).
__device__ __forceinline__ void fill_b_swz(__bf16* swz, const float* W, int ld,
                                           int kchunks, int t, int nthreads) {
  const int items = kchunks * 16 * 32;  // (frag, lane) pairs
  for (int i = t; i < items; i += nthreads) {
    const int lane = i & 31;
    const int f = i >> 5;
    const int kc = f >> 4;
    const int nt = f & 15;
    const int n = nt * 16 + (lane & 15);
    const int k0 = kc * 32 + (lane >> 4) * 16;
    __bf16* dst = swz + (size_t)i * 16;
#pragma unroll
    for (int e = 0; e < 16; ++e) dst[e] = (__bf16)W[(size_t)(k0 + e) * ld + n];
  }
}

// ======================= Kernel 1: input projection =======================
// grid = SEQ (one time-step each), 256 threads (8 waves).
__global__ void xproj_kernel(const int* __restrict__ x,
                             const float* __restrict__ embedding,
                             const float* __restrict__ W_xh,
                             const float* __restrict__ b_h,
                             float* __restrict__ xproj /* [S*B, H] */) {
  extern __shared__ __bf16 smem1[];
  __bf16* aLds = smem1;                 // [128][EMBED] gathered embedding rows
  __bf16* wSwz = smem1 + 128 * EMBED;   // 4 kc x 16 nt swizzled frags (64 KB)

  const int s = blockIdx.x;
  const int t = threadIdx.x;

  {  // gather + convert embedding rows (float4-vectorized)
    const int r  = t >> 1;
    const int c0 = (t & 1) * 64;
    const int ev = x[r * SEQ + s];
    const float4* er4 = (const float4*)(embedding + (size_t)ev * EMBED + c0);
#pragma unroll
    for (int j = 0; j < 16; ++j)
      cvt_store4(aLds + r * EMBED + c0 + j * 4, er4[j]);
  }
  fill_b_swz(wSwz, W_xh, HIDDEN, /*kchunks=*/4, t, 256);
  __syncthreads();

  const int wave = t >> 5;
  const int lane = t & 31;

  v16bf afrag[4];
#pragma unroll
  for (int kc = 0; kc < 4; ++kc)
    afrag[kc] = load_a_frag(aLds + (wave * 16) * EMBED + kc * 32, EMBED, lane);

  const int nloc = lane & 15;
  const int mofs = (lane >> 4) * 8;
  const size_t rowbase = (size_t)s * BATCH + wave * 16;

  for (int nt = 0; nt < 16; ++nt) {
    v8f c = {};
#pragma unroll
    for (int kc = 0; kc < 4; ++kc) {
      v16bf b = load_b_swz(wSwz, kc * 16 + nt, lane);
      c = __builtin_amdgcn_wmma_f32_16x16x32_bf16(false, afrag[kc], false, b,
                                                  (short)0, c, false, false);
    }
    const int n = nt * 16 + nloc;
    const float bias = b_h[n];
#pragma unroll
    for (int r = 0; r < 8; ++r)
      xproj[(rowbase + mofs + r) * HIDDEN + n] = c[r] + bias;
  }
}

// ======================= Kernel 2: recurrence =======================
// grid = BATCH/16 (independent batch slices), 512 threads = 16 waves,
// wave = one 16-col N-tile of H. W_hh B-fragments live in VGPRs for the
// whole sequence; double-buffered h tile in LDS; one barrier per step.
__global__ void rnn_step_kernel(const float* __restrict__ hidden0,
                                const float* __restrict__ W_hh,
                                float* __restrict__ xh /* in: xproj, out: hs */,
                                float* __restrict__ h_last_out /* [B,H] */) {
  extern __shared__ __bf16 smem2[];
  __bf16* wSwz = smem2;                         // 8 kc x 16 nt frags (128 KB)
  __bf16* hbuf = smem2 + 8 * 16 * 32 * 16;      // [2][16][H] bf16 (16 KB)

  const int bb   = blockIdx.x * 16;
  const int t    = threadIdx.x;
  const int wave = t >> 5;
  const int lane = t & 31;

  fill_b_swz(wSwz, W_hh, HIDDEN, /*kchunks=*/8, t, 512);
  for (int i = t * 4; i < 16 * HIDDEN; i += 512 * 4) {  // h_0 -> buffer 0
    const int m = i >> 8, k = i & (HIDDEN - 1);
    cvt_store4(hbuf + i, *(const float4*)(hidden0 + (size_t)(bb + m) * HIDDEN + k));
  }
  __syncthreads();

  // hoist this wave's 8 W_hh B-fragments into registers for all 1024 steps
  v16bf bfrag[8];
#pragma unroll
  for (int kc = 0; kc < 8; ++kc)
    bfrag[kc] = load_b_swz(wSwz, kc * 16 + wave, lane);

  const int nloc = lane & 15;
  const int mofs = (lane >> 4) * 8;
  const int n    = wave * 16 + nloc;

  for (int s = 0; s < SEQ; ++s) {
    const __bf16* hin  = hbuf + (s & 1) * (16 * HIDDEN);
    __bf16*       hout = hbuf + ((s + 1) & 1) * (16 * HIDDEN);

    // batch-issue all A-fragment DS loads, then run the WMMA chain
    v16bf a[8];
#pragma unroll
    for (int kc = 0; kc < 8; ++kc)
      a[kc] = load_a_frag(hin + kc * 32, HIDDEN, lane);

    v8f c = {};
#pragma unroll
    for (int kc = 0; kc < 8; ++kc)
      c = __builtin_amdgcn_wmma_f32_16x16x32_bf16(false, a[kc], false, bfrag[kc],
                                                  (short)0, c, false, false);

    float hv[8];
    const float* xp = xh + ((size_t)s * BATCH + bb) * HIDDEN;
#pragma unroll
    for (int r = 0; r < 8; ++r)
      hv[r] = tanhf(c[r] + xp[(size_t)(mofs + r) * HIDDEN + n]);

    if (s + 1 < SEQ)  // pull next step's xproj toward the caches
      __builtin_prefetch(xh + ((size_t)(s + 1) * BATCH + bb + mofs) * HIDDEN + n, 0, 1);

#pragma unroll
    for (int r = 0; r < 8; ++r) {
      const int m = mofs + r;
      hout[m * HIDDEN + n] = (__bf16)hv[r];                     // h for step s+1
      xh[((size_t)s * BATCH + bb + m) * HIDDEN + n] = hv[r];    // hs (in place)
    }
    if (s == SEQ - 1) {
#pragma unroll
      for (int r = 0; r < 8; ++r)
        h_last_out[(size_t)(bb + mofs + r) * HIDDEN + n] = hv[r];
    }
    __syncthreads();  // single barrier: h_s visible before step s+1 reads
  }
}

// ======================= Kernel 3: output projection =======================
// grid = SEQ (one time-step -> 128 contiguous hs rows), 256 threads.
__global__ void logits_kernel(const float* __restrict__ hs /* [S*B,H] */,
                              const float* __restrict__ W_hy,
                              const float* __restrict__ b_y,
                              float* __restrict__ out /* [B,S,V] */) {
  extern __shared__ __bf16 smem3[];
  __bf16* aLds = smem3;                  // [128][H] (64 KB)
  __bf16* wSwz = smem3 + 128 * HIDDEN;   // 8 kc x 16 nt frags (128 KB)

  const int s = blockIdx.x;
  const int t = threadIdx.x;

  const float* arows = hs + (size_t)s * BATCH * HIDDEN;
  for (int i = t * 4; i < 128 * HIDDEN; i += 256 * 4)
    cvt_store4(aLds + i, *(const float4*)(arows + i));
  fill_b_swz(wSwz, W_hy, VOCAB, /*kchunks=*/8, t, 256);
  __syncthreads();

  const int wave = t >> 5;
  const int lane = t & 31;

  v16bf afrag[8];
#pragma unroll
  for (int kc = 0; kc < 8; ++kc)
    afrag[kc] = load_a_frag(aLds + (wave * 16) * HIDDEN + kc * 32, HIDDEN, lane);

  const int nloc = lane & 15;
  const int mofs = (lane >> 4) * 8;

  for (int nt = 0; nt < 16; ++nt) {
    v8f c = {};
#pragma unroll
    for (int kc = 0; kc < 8; ++kc) {
      v16bf b = load_b_swz(wSwz, kc * 16 + nt, lane);
      c = __builtin_amdgcn_wmma_f32_16x16x32_bf16(false, afrag[kc], false, b,
                                                  (short)0, c, false, false);
    }
    const int n = nt * 16 + nloc;
    const float bias = b_y[n];
#pragma unroll
    for (int r = 0; r < 8; ++r) {
      const int bidx = wave * 16 + mofs + r;                   // batch row
      out[((size_t)bidx * SEQ + s) * VOCAB + n] = c[r] + bias; // [B,S,V]
    }
  }
}

extern "C" void kernel_launch(void* const* d_in, const int* in_sizes, int n_in,
                              void* d_out, int out_size, void* d_ws, size_t ws_size,
                              hipStream_t stream) {
  const int*   x         = (const int*)  d_in[0];
  const float* hidden    = (const float*)d_in[1];
  const float* embedding = (const float*)d_in[2];
  const float* W_xh      = (const float*)d_in[3];
  const float* W_hh      = (const float*)d_in[4];
  const float* b_h       = (const float*)d_in[5];
  const float* W_hy      = (const float*)d_in[6];
  const float* b_y       = (const float*)d_in[7];
  float* out = (float*)d_out;
  float* ws  = (float*)d_ws;   // xproj -> hs, [S*B*H] f32 = 128 MiB (L2-resident)

  const size_t lds1 = (size_t)(128 * EMBED + 4 * 16 * 32 * 16) * sizeof(__bf16);     //  96 KB
  const size_t lds2 = (size_t)(8 * 16 * 32 * 16 + 2 * 16 * HIDDEN) * sizeof(__bf16); // 144 KB
  const size_t lds3 = (size_t)(128 * HIDDEN + 8 * 16 * 32 * 16) * sizeof(__bf16);    // 192 KB

  float* h_last_out = out + (size_t)BATCH * SEQ * VOCAB;

  xproj_kernel   <<<SEQ,        256, lds1, stream>>>(x, embedding, W_xh, b_h, ws);
  rnn_step_kernel<<<BATCH / 16, 512, lds2, stream>>>(hidden, W_hh, ws, h_last_out);
  logits_kernel  <<<SEQ,        256, lds3, stream>>>(ws, W_hy, b_y, out);
}
